// GCN_28063316312557
// MI455X (gfx1250) — matrix operand order repaired and checked by reading, not compiled, not run on hardware
//
#include <hip/hip_runtime.h>
#include <hip/hip_fp16.h>

typedef __attribute__((ext_vector_type(16))) _Float16 v16h;
typedef __attribute__((ext_vector_type(8)))  float    v8f;

#define F_IN 128
#define HID  16
#define NCLS 40

// ---------------------------------------------------------------------------
// degree: deg[dst] += 1 per edge (self-loop +1 added in dinv kernel)
// ---------------------------------------------------------------------------
__global__ void deg_kernel(const long long* __restrict__ ei, float* __restrict__ deg, int E) {
    int e = blockIdx.x * blockDim.x + threadIdx.x;
    if (e >= E) return;
    int dst = (int)ei[(size_t)E + e];
    atomicAdd(&deg[dst], 1.0f);
}

__global__ void dinv_kernel(const float* __restrict__ deg, float* __restrict__ dinv, int N) {
    int i = blockIdx.x * blockDim.x + threadIdx.x;
    if (i >= N) return;
    dinv[i] = rsqrtf(deg[i] + 1.0f);
}

// ---------------------------------------------------------------------------
// GEMM1: h = x @ W1   (M=100000 [%16==0], K=128, N=16)  -- WMMA f16->f32
// One wave per 16-row tile; K loop = 4 x v_wmma_f32_16x16x32_f16.
// ---------------------------------------------------------------------------
__global__ void gemm1_wmma(const float* __restrict__ x, const float* __restrict__ W1,
                           float* __restrict__ h, int ntiles) {
    int wave = threadIdx.x >> 5;
    int lane = threadIdx.x & 31;
    int tile = blockIdx.x * (blockDim.x >> 5) + wave;
    if (tile >= ntiles) return;                 // uniform per-wave exit: EXEC all-1s at WMMA

    int mrow = (tile << 4) + (lane & 15);
    int kg   = lane >> 4;                        // 0: lanes 0-15, 1: lanes 16-31
    int col  = lane & 15;

    const float* xrow = x + (size_t)mrow * F_IN;
    v8f c = {};

#pragma unroll
    for (int ks = 0; ks < 4; ++ks) {
        int k0 = ks * 32;
        // A fragment (16-bit A 16x32 layout): halves 0..7 = K=k0+kg*8+j,
        //                                     halves 8..15 = K=k0+16+kg*8+j
        const float4* pa = (const float4*)(xrow + k0 + kg * 8);
        const float4* pb = (const float4*)(xrow + k0 + 16 + kg * 8);
        float4 a0 = pa[0], a1 = pa[1];
        float4 b0 = pb[0], b1 = pb[1];
        v16h A;
        A[0]  = (_Float16)a0.x; A[1]  = (_Float16)a0.y; A[2]  = (_Float16)a0.z; A[3]  = (_Float16)a0.w;
        A[4]  = (_Float16)a1.x; A[5]  = (_Float16)a1.y; A[6]  = (_Float16)a1.z; A[7]  = (_Float16)a1.w;
        A[8]  = (_Float16)b0.x; A[9]  = (_Float16)b0.y; A[10] = (_Float16)b0.z; A[11] = (_Float16)b0.w;
        A[12] = (_Float16)b1.x; A[13] = (_Float16)b1.y; A[14] = (_Float16)b1.z; A[15] = (_Float16)b1.w;

        // B fragment (32x16, row striped across lanes): lane = column,
        // halves 0..7 = K=k0+kg*8+j, halves 8..15 = K=k0+16+kg*8+j (matches A).
        v16h B;
#pragma unroll
        for (int j = 0; j < 16; ++j)
            B[j] = (_Float16)W1[(size_t)(k0 + kg * 8 + j + (j >= 8 ? 8 : 0)) * HID + col];

        c = __builtin_amdgcn_wmma_f32_16x16x32_f16(
                /*neg_a=*/false, A, /*neg_b=*/false, B,
                /*c_mod=*/(short)0, c, /*reuse_a=*/false, /*reuse_b=*/false);
    }

    // C/D layout: VGPR r -> row r (lanes 0-15) / row r+8 (lanes 16-31), col = lane&15
    int rbase = kg * 8;
#pragma unroll
    for (int r = 0; r < 8; ++r)
        h[(size_t)((tile << 4) + rbase + r) * HID + col] = c[r];
}

// ---------------------------------------------------------------------------
// Edge aggregation (16 lanes per edge, one feature each):
//   agg[dst][f] += feat[src][f] * dinv[src] * dinv[dst]
// Feature tables (6.4 MB) are L2-resident; atomics resolve in L2.
// ---------------------------------------------------------------------------
__global__ void edge_agg_kernel(const long long* __restrict__ ei,
                                const float* __restrict__ feat,
                                const float* __restrict__ dinv,
                                float* __restrict__ agg, int E) {
    long long gid = (long long)blockIdx.x * blockDim.x + threadIdx.x;
    if (gid >= (long long)E * HID) return;
    int e = (int)(gid >> 4);
    int f = (int)(gid & 15);
    __builtin_prefetch(&ei[e + 2048], 0, 0);     // global_prefetch_b8 on the edge stream
    int src = (int)ei[e];
    int dst = (int)ei[(size_t)E + e];
    float norm = dinv[src] * dinv[dst];
    atomicAdd(&agg[(size_t)dst * HID + f], feat[(size_t)src * HID + f] * norm);
}

// ---------------------------------------------------------------------------
// Layer-1 epilogue: h1 = relu(agg1 + h*dinv^2 + b1), stored in-place over agg1.
// Also zeroes the h buffer so it can be reused as the agg2 accumulator.
// ---------------------------------------------------------------------------
__global__ void selfloop_relu_kernel(const float* __restrict__ dinv,
                                     float* __restrict__ agg1_then_h1,
                                     float* __restrict__ h_then_agg2,
                                     const float* __restrict__ b1, int N) {
    int gid = blockIdx.x * blockDim.x + threadIdx.x;
    if (gid >= N * HID) return;
    int i = gid >> 4;
    int f = gid & 15;
    float di = dinv[i];
    float v = agg1_then_h1[gid] + h_then_agg2[gid] * di * di + b1[f];
    agg1_then_h1[gid] = fmaxf(v, 0.0f);   // -> h1
    h_then_agg2[gid]  = 0.0f;             // -> zeroed agg2 accumulator
}

// ---------------------------------------------------------------------------
// Layer-2 tail: z = agg2 + h1*dinv^2 ; logits = z @ W2 + b2 ; log_softmax.
// (Uses  Â(h1 W2) = (Â h1) W2  so the edge pass stayed 16-wide.)
// ---------------------------------------------------------------------------
__global__ void out_kernel(const float* __restrict__ agg2, const float* __restrict__ h1,
                           const float* __restrict__ dinv, const float* __restrict__ W2,
                           const float* __restrict__ b2, float* __restrict__ out, int N) {
    __shared__ float sW[HID * NCLS];
    __shared__ float sb[NCLS];
    for (int t = threadIdx.x; t < HID * NCLS; t += blockDim.x) sW[t] = W2[t];
    for (int t = threadIdx.x; t < NCLS; t += blockDim.x) sb[t] = b2[t];
    __syncthreads();

    int i = blockIdx.x * blockDim.x + threadIdx.x;
    if (i >= N) return;

    float di = dinv[i];
    float dii = di * di;
    float z[HID];
#pragma unroll
    for (int f = 0; f < HID; ++f)
        z[f] = agg2[(size_t)i * HID + f] + h1[(size_t)i * HID + f] * dii;

    float l[NCLS];
    float m = -1e30f;
#pragma unroll 4
    for (int c = 0; c < NCLS; ++c) {
        float acc = sb[c];
#pragma unroll
        for (int f = 0; f < HID; ++f) acc += z[f] * sW[f * NCLS + c];
        l[c] = acc;
        m = fmaxf(m, acc);
    }
    float s = 0.0f;
#pragma unroll 4
    for (int c = 0; c < NCLS; ++c) s += __expf(l[c] - m);
    float ls = __logf(s);
#pragma unroll 4
    for (int c = 0; c < NCLS; ++c) out[(size_t)i * NCLS + c] = l[c] - m - ls;
}

// ---------------------------------------------------------------------------
extern "C" void kernel_launch(void* const* d_in, const int* in_sizes, int n_in,
                              void* d_out, int out_size, void* d_ws, size_t ws_size,
                              hipStream_t stream) {
    const float*     x  = (const float*)d_in[0];
    const long long* ei = (const long long*)d_in[1];   // int64 [2, E]
    const float*     W1 = (const float*)d_in[2];
    const float*     b1 = (const float*)d_in[3];
    const float*     W2 = (const float*)d_in[4];
    const float*     b2 = (const float*)d_in[5];
    float* out = (float*)d_out;

    int N = in_sizes[0] / F_IN;   // 100000 (divisible by 16 -> no WMMA tail)
    int E = in_sizes[1] / 2;      // 3200000

    // workspace layout (~13.7 MB)
    float* deg  = (float*)d_ws;                 // [N]
    float* dinv = deg + N;                      // [N]
    float* bufA = dinv + N;                     // [N*HID]  h  -> agg2
    float* bufB = bufA + (size_t)N * HID;       // [N*HID]  agg1 -> h1

    hipMemsetAsync(deg,  0, (size_t)N * sizeof(float), stream);
    hipMemsetAsync(bufB, 0, (size_t)N * HID * sizeof(float), stream);

    deg_kernel <<<(E + 255) / 256, 256, 0, stream>>>(ei, deg, E);
    dinv_kernel<<<(N + 255) / 256, 256, 0, stream>>>(deg, dinv, N);

    int ntiles = (N + 15) / 16;                 // 6250
    int gblocks = (ntiles + 7) / 8;             // 8 waves / block
    gemm1_wmma<<<gblocks, 256, 0, stream>>>(x, W1, bufA, ntiles);

    long long ework = (long long)E * HID;
    int eblocks = (int)((ework + 255) / 256);
    edge_agg_kernel<<<eblocks, 256, 0, stream>>>(ei, bufA, dinv, bufB, E);   // agg1 += h[src]*norm

    selfloop_relu_kernel<<<(N * HID + 255) / 256, 256, 0, stream>>>(dinv, bufB, bufA, b1, N);

    edge_agg_kernel<<<eblocks, 256, 0, stream>>>(ei, bufB, dinv, bufA, E);   // agg2 += h1[src]*norm

    out_kernel<<<(N + 255) / 256, 256, 0, stream>>>(bufA, bufB, dinv, W2, b2, out, N);
}
